// SelfAttentionLayer_56453050139024
// MI455X (gfx1250) — matrix operand order repaired and checked by reading, not compiled, not run on hardware
//
#include <hip/hip_runtime.h>
#include <hip/hip_bf16.h>

typedef __attribute__((ext_vector_type(16))) _Float16 v16h;
typedef __attribute__((ext_vector_type(8)))  _Float16 v8h;
typedef __attribute__((ext_vector_type(8)))  float    v8f;
typedef __attribute__((ext_vector_type(4)))  float    v4f;

#define AN               9216   // 96*96 positions
#define NBATCH           2
#define WAVES_PER_BLOCK  8
#define TILE_M           16
#define TILE_K           32
#define ATTN_SCALE       0.125f // 64^-0.5
#define LOG2E            1.44269504088896340736f

// ---------------------------------------------------------------------------
// Kernel 1: per-batch min/max of x -> ws[b] = max, ws[NBATCH+b] = min.
// Rank-1 logits => softmax row-max is a_i*kmax or a_i*kmin; this tiny
// reduction replaces the whole N^2 max pass.
// ---------------------------------------------------------------------------
__global__ void __launch_bounds__(256)
sa_minmax_kernel(const float* __restrict__ x, float* __restrict__ ws) {
  __shared__ float smx[256];
  __shared__ float smn[256];
  const int b = blockIdx.x;
  const float* xb = x + b * AN;
  float mx = -3.4e38f, mn = 3.4e38f;
  for (int i = threadIdx.x; i < AN; i += 256) {
    const float v = xb[i];
    mx = fmaxf(mx, v);
    mn = fminf(mn, v);
  }
  smx[threadIdx.x] = mx;
  smn[threadIdx.x] = mn;
  __syncthreads();
  for (int s = 128; s > 0; s >>= 1) {
    if (threadIdx.x < s) {
      smx[threadIdx.x] = fmaxf(smx[threadIdx.x], smx[threadIdx.x + s]);
      smn[threadIdx.x] = fminf(smn[threadIdx.x], smn[threadIdx.x + s]);
    }
    __syncthreads();
  }
  if (threadIdx.x == 0) {
    ws[b]          = smx[0];
    ws[NBATCH + b] = smn[0];
  }
}

// ---------------------------------------------------------------------------
// Kernel 2: fused rank-1 attention. One wave per 16-row tile; per 32-col
// chunk one v_wmma_f32_16x16x32_f16 accumulates numerator (B col 0 = v_j)
// and denominator (B col 1 = 1) of the softmax reduction in f32.
// exp2-folded exponent (1 FMA + 1 v_exp per element), j-loop unrolled x2
// with independent accumulators so exp streams and XDL ops overlap.
// ---------------------------------------------------------------------------
__global__ void __launch_bounds__(256)
sa_attn_kernel(const float* __restrict__ x, const float* __restrict__ wqkv,
               const float* __restrict__ wout, const float* __restrict__ bout,
               const float* __restrict__ minmax, float* __restrict__ out) {
  __shared__ __align__(16) float    s_x[AN];         // 36 KB: raw x (A path)
  __shared__ __align__(16) _Float16 s_vh[AN];        // 18 KB: v_j = x_j*wv, f16
  __shared__ float s_nd[WAVES_PER_BLOCK][32];        // num[0..15], den[16..31]

  const int b = blockIdx.y;
  const float* xb = x + b * AN;
  __builtin_prefetch(xb, 0, 3);                      // global_prefetch_b8

  const float wq = wqkv[0], wk = wqkv[1], wv = wqkv[2];
  for (int i = threadIdx.x; i < AN; i += 256) {
    const float xi = xb[i];
    s_x[i]  = xi;
    s_vh[i] = (_Float16)(xi * wv);
  }
  __syncthreads();

  const int  wave = threadIdx.x >> 5;
  const int  lane = threadIdx.x & 31;
  const int  m    = lane & 15;                 // A row within tile / B,D column
  const bool hi   = lane >= 16;
  const int  tile = blockIdx.x * WAVES_PER_BLOCK + wave;
  const int  i0   = tile * TILE_M;

  const float xmax = minmax[b], xmin = minmax[NBATCH + b];
  const float kmx = (wk >= 0.f) ? wk * xmax : wk * xmin;   // max_j k_j
  const float kmn = (wk >= 0.f) ? wk * xmin : wk * xmax;   // min_j k_j

  const float a    = s_x[i0 + m] * wq * ATTN_SCALE;        // a_i for my row
  const float rmax = (a >= 0.f) ? a * kmx : a * kmn;       // exact row max
  // exp2 folding: e^(a*wk*x - rmax) = 2^(c1*x + c0)
  const float c1 = a * wk * LOG2E;
  const float c0 = -rmax * LOG2E;

  // Per-lane loop-invariant B selectors: col0 = v_j, col1 = 1, rest 0.
  const _Float16 s0 = (_Float16)((m == 0) ? 1.0f : 0.0f);
  const _Float16 s1 = (_Float16)((m == 1) ? 1.0f : 0.0f);

  // 16-bit A 16x32 layout: lanes 0-15 hold K {0..7, 16..23}; lanes 16-31 hold
  // K {8..15, 24..31}.  B 32x16: same K split as A.
  const int akoff = hi ? 8 : 0;
  const int bkoff = hi ? 8 : 0;

  const v4f* px = (const v4f*)s_x;    // j0,akoff multiples of 8 -> /4 exact
  const v8h* pv = (const v8h*)s_vh;   // j0,bkoff multiples of 8 -> /8 exact

  v8f acc0 = {}, acc1 = {};
  for (int j0 = 0; j0 < AN; j0 += 2 * TILE_K) {      // 144 iterations
#pragma unroll
    for (int u = 0; u < 2; ++u) {
      const int jc = j0 + u * TILE_K;

      // ---- A: exp2(c1*x_j + c0), two runs of 8 contiguous floats ----
      const int ab = (jc + akoff) >> 2;
      const v4f x0 = px[ab], x1 = px[ab + 1], x2 = px[ab + 4], x3 = px[ab + 5];
      v16h Am;
#pragma unroll
      for (int h = 0; h < 4; ++h) {
        Am[h]      = (_Float16)__builtin_amdgcn_exp2f(fmaf(c1, x0[h], c0));
        Am[4 + h]  = (_Float16)__builtin_amdgcn_exp2f(fmaf(c1, x1[h], c0));
        Am[8 + h]  = (_Float16)__builtin_amdgcn_exp2f(fmaf(c1, x2[h], c0));
        Am[12 + h] = (_Float16)__builtin_amdgcn_exp2f(fmaf(c1, x3[h], c0));
      }

      // ---- B: branchless v_h*s0 + s1, two runs of 8 contiguous halfs ----
      const int bb = (jc + bkoff) >> 3;
      const v8h v0 = pv[bb], v1 = pv[bb + 2];
      v16h Bm;
#pragma unroll
      for (int h = 0; h < 8; ++h) {
        Bm[h]     = (_Float16)(v0[h] * s0 + s1);
        Bm[8 + h] = (_Float16)(v1[h] * s0 + s1);
      }

      // D[:,0] += E*v (numerator), D[:,1] += E*1 (denominator)
      if (u == 0)
        acc0 = __builtin_amdgcn_wmma_f32_16x16x32_f16(
            false, Am, false, Bm, (short)0, acc0, false, false);
      else
        acc1 = __builtin_amdgcn_wmma_f32_16x16x32_f16(
            false, Am, false, Bm, (short)0, acc1, false, false);
    }
  }
  const v8f acc = acc0 + acc1;

  // D layout: lane holds column n = m; acc[g] = D[M = g + (hi?8:0)][n].
  if (m < 2) {
    const int mrow = hi ? 8 : 0;
#pragma unroll
    for (int g = 0; g < 8; ++g)
      s_nd[wave][m * 16 + mrow + g] = acc[g];
  }
  __syncthreads();

  if (lane < 16) {
    const float num = s_nd[wave][lane];
    const float den = s_nd[wave][16 + lane];
    const float r   = num * __builtin_amdgcn_rcpf(den);    // softmax normalize
    out[b * AN + i0 + lane] = r * wout[0] + bout[0];
  }
}

// ---------------------------------------------------------------------------
extern "C" void kernel_launch(void* const* d_in, const int* in_sizes, int n_in,
                              void* d_out, int out_size, void* d_ws, size_t ws_size,
                              hipStream_t stream) {
  const float* x    = (const float*)d_in[0];   // [2,1,96,96]
  const float* wqkv = (const float*)d_in[1];   // [3,1,1,1]
  const float* wout = (const float*)d_in[2];   // [1,1,1,1]
  const float* bout = (const float*)d_in[3];   // [1]
  float* out = (float*)d_out;                  // [2,1,96,96] f32
  float* ws  = (float*)d_ws;                   // 4 floats: max[2], min[2]

  sa_minmax_kernel<<<dim3(NBATCH), dim3(256), 0, stream>>>(x, ws);

  dim3 grid(AN / TILE_M / WAVES_PER_BLOCK, NBATCH);  // (72, 2)
  sa_attn_kernel<<<grid, dim3(256), 0, stream>>>(x, wqkv, wout, bout, ws, out);
}
